// DeepseekV2DecoderLayer_52785148067910
// MI455X (gfx1250) — matrix operand order repaired and checked by reading, not compiled, not run on hardware
//
#include <hip/hip_runtime.h>
#include <hip/hip_bf16.h>
#include <math.h>

#define T_TOK 2048
#define HID   2048
#define NH    16
#define DN    128
#define DR    64
#define DQK   192
#define DV    128
#define QLR   1536
#define KVLR  512
#define INTER 10944
#define EPS_RMS 1e-6f

typedef __bf16 bf16_t;
typedef __attribute__((ext_vector_type(16))) __bf16 bf16x16;
typedef __attribute__((ext_vector_type(8)))  float  f32x8;
typedef __attribute__((__vector_size__(16))) int v4i_t;  // async-LDS operand type

#if __has_builtin(__builtin_amdgcn_global_load_async_to_lds_b128)
#define ASYNC_LDS 1
#else
#define ASYNC_LDS 0
#endif

// ---------------------------------------------------------------------------
// CDNA5 WMMA fragment loaders (wave32; A/B use this file's contiguous k order, same in both).
// A 16x32 bf16: lane L(0-15) row=L, kbase=0; lane L(16-31) row=L-16, kbase=16.
//   halves 0..15 -> K = kbase + 0..15       (32 contiguous bytes)
// B 32x16 bf16: lane L: col N = L&15, kbase = (L<16)?0:16,
//   halves 0..15 -> K = kbase + 0..15       (32 contiguous bytes)
// C/D f32 16x16: lane L: N = L&15, VGPR j -> M = j + ((L<16)?0:8)
// ---------------------------------------------------------------------------
static __device__ inline bf16x16 ld_frag_a(const bf16_t* p) {
  bf16x16 f;
  reinterpret_cast<uint4*>(&f)[0] = *reinterpret_cast<const uint4*>(p);
  reinterpret_cast<uint4*>(&f)[1] = *reinterpret_cast<const uint4*>(p + 8);
  return f;
}
static __device__ inline bf16x16 ld_frag_b(const bf16_t* p) {
  bf16x16 f;
  reinterpret_cast<uint4*>(&f)[0] = *reinterpret_cast<const uint4*>(p);
  reinterpret_cast<uint4*>(&f)[1] = *reinterpret_cast<const uint4*>(p + 8);
  return f;
}
static __device__ inline f32x8 wmma_bf16(bf16x16 a, bf16x16 b, f32x8 c) {
  return __builtin_amdgcn_wmma_f32_16x16x32_bf16(false, a, false, b, (short)0, c,
                                                 false, false);
}

// ---------------------------------------------------------------------------
// Generic NT GEMM: OUT[M,N] = A[M,K] * W[N,K]^T, fp32 in, bf16 WMMA, fp32 acc.
// Block: 256 thr = 8 waves; tile 256x64; wave tile 64x32 (4x2 WMMA).
// Double-buffered LDS; next K-tile staged through registers during WMMA.
// EPI: 0 = store; 1 = store + X (residual); 2 = silu(X) * result (MLP fuse).
// ---------------------------------------------------------------------------
template <int EPI>
__global__ void __launch_bounds__(256)
gemm_bf16wmma(const float* __restrict__ A, int lda,
              const float* __restrict__ W, int ldw,
              float* __restrict__ O, int ldo,
              const float* __restrict__ X, int ldx, int K) {
  __shared__ bf16_t As[2][256][32];
  __shared__ bf16_t Bs[2][64][32];
  const int tid = threadIdx.x;
  const int mBase = blockIdx.y * 256;
  const int nBase = blockIdx.x * 64;
  const int wave = tid >> 5, lane = tid & 31;
  const int wm = (wave & 3) * 64, wn = (wave >> 2) * 32;
  const int hi = lane >> 4, l15 = lane & 15;

  const f32x8 zero8 = {0.f, 0.f, 0.f, 0.f, 0.f, 0.f, 0.f, 0.f};
  f32x8 acc[4][2];
#pragma unroll
  for (int mi = 0; mi < 4; ++mi)
#pragma unroll
    for (int ni = 0; ni < 2; ++ni) acc[mi][ni] = zero8;

  float4 aR[8], bR[2];  // register staging for the next K-tile

  auto loadTile = [&](int k0) {
#pragma unroll
    for (int i = 0; i < 8; ++i) {  // A: 256x32 fp32 = 2048 float4
      int f4 = tid + 256 * i;
      int r = f4 >> 3, c4 = (f4 & 7) * 4;
      aR[i] = *reinterpret_cast<const float4*>(A + (size_t)(mBase + r) * lda +
                                               k0 + c4);
    }
#pragma unroll
    for (int i = 0; i < 2; ++i) {  // W: 64x32 fp32 = 512 float4
      int f4 = tid + 256 * i;
      int r = f4 >> 3, c4 = (f4 & 7) * 4;
      bR[i] = *reinterpret_cast<const float4*>(W + (size_t)(nBase + r) * ldw +
                                               k0 + c4);
    }
  };
  auto storeTile = [&](int buf) {
#pragma unroll
    for (int i = 0; i < 8; ++i) {
      int f4 = tid + 256 * i;
      int r = f4 >> 3, c4 = (f4 & 7) * 4;
      bf16_t t[4] = {(bf16_t)aR[i].x, (bf16_t)aR[i].y, (bf16_t)aR[i].z,
                     (bf16_t)aR[i].w};
      *reinterpret_cast<uint2*>(&As[buf][r][c4]) =
          *reinterpret_cast<uint2*>(t);
    }
#pragma unroll
    for (int i = 0; i < 2; ++i) {
      int f4 = tid + 256 * i;
      int r = f4 >> 3, c4 = (f4 & 7) * 4;
      bf16_t t[4] = {(bf16_t)bR[i].x, (bf16_t)bR[i].y, (bf16_t)bR[i].z,
                     (bf16_t)bR[i].w};
      *reinterpret_cast<uint2*>(&Bs[buf][r][c4]) =
          *reinterpret_cast<uint2*>(t);
    }
  };

  const int nk = K / 32;
  loadTile(0);
  storeTile(0);
  __syncthreads();

  for (int kt = 0; kt < nk; ++kt) {
    const int cur = kt & 1;
    if (kt + 1 < nk) loadTile((kt + 1) * 32);  // global loads overlap WMMA

    bf16x16 af[4], bfg[2];
#pragma unroll
    for (int mi = 0; mi < 4; ++mi)
      af[mi] = ld_frag_a(&As[cur][wm + mi * 16 + l15][hi * 16]);
#pragma unroll
    for (int ni = 0; ni < 2; ++ni)
      bfg[ni] = ld_frag_b(&Bs[cur][wn + ni * 16 + l15][hi * 16]);
#pragma unroll
    for (int mi = 0; mi < 4; ++mi)
#pragma unroll
      for (int ni = 0; ni < 2; ++ni)
        acc[mi][ni] = wmma_bf16(af[mi], bfg[ni], acc[mi][ni]);

    if (kt + 1 < nk) storeTile(cur ^ 1);
    __syncthreads();
  }

#pragma unroll
  for (int mi = 0; mi < 4; ++mi)
#pragma unroll
    for (int ni = 0; ni < 2; ++ni)
#pragma unroll
      for (int j = 0; j < 8; ++j) {
        int row = mBase + wm + mi * 16 + j + hi * 8;
        int col = nBase + wn + ni * 16 + l15;
        float v = acc[mi][ni][j];
        size_t o = (size_t)row * ldo + col;
        if (EPI == 0) {
          O[o] = v;
        } else if (EPI == 1) {
          O[o] = v + X[(size_t)row * ldx + col];
        } else {
          float g = X[(size_t)row * ldx + col];
          O[o] = (g / (1.f + __expf(-g))) * v;  // silu(gate) * up
        }
      }
}

// ---------------------------------------------------------------------------
// RMSNorm, one block per row; supports in-place (y == x) and strided rows.
// ---------------------------------------------------------------------------
__global__ void __launch_bounds__(256)
rmsnorm_k(const float* __restrict__ x, const float* __restrict__ w,
          float* __restrict__ y, int D, int sx, int sy) {
  __shared__ float red[8];
  const int t = blockIdx.x, tid = threadIdx.x;
  const float* xr = x + (size_t)t * sx;
  float* yr = y + (size_t)t * sy;
  float v[8];
  float ss = 0.f;
  int n = 0;
  for (int i = tid; i < D; i += 256) {
    float f = xr[i];
    v[n++] = f;
    ss += f * f;
  }
#pragma unroll
  for (int m = 16; m >= 1; m >>= 1) ss += __shfl_xor(ss, m, 32);
  if ((tid & 31) == 0) red[tid >> 5] = ss;
  __syncthreads();
  float tot = 0.f;
#pragma unroll
  for (int i = 0; i < 8; ++i) tot += red[i];
  const float r = rsqrtf(tot / (float)D + EPS_RMS);
  n = 0;
  for (int i = tid; i < D; i += 256) yr[i] = v[n++] * r * w[i];
}

// ---------------------------------------------------------------------------
// RoPE (deinterleave + rotate-half) and pack Q/K/V to bf16.
// ---------------------------------------------------------------------------
__global__ void __launch_bounds__(256)
rope_pack_k(const float* __restrict__ qb, const float* __restrict__ ckv,
            const float* __restrict__ kvb, const float* __restrict__ cosT,
            const float* __restrict__ sinT, bf16_t* __restrict__ Qf,
            bf16_t* __restrict__ Kf, bf16_t* __restrict__ Vb) {
  __shared__ float kemb[DR];
  const int t = blockIdx.x, tid = threadIdx.x;
  const float* cs = cosT + (size_t)t * DR;
  const float* sn = sinT + (size_t)t * DR;
  if (tid < DR) {
    const float* kp = ckv + (size_t)t * (KVLR + DR) + KVLR;
    int i = tid;
    float y, rt;
    if (i < 32) { y = kp[2 * i];           rt = -kp[2 * i + 1]; }
    else        { int j = i - 32; y = kp[2 * j + 1]; rt = kp[2 * j]; }
    kemb[i] = y * cs[i] + rt * sn[i];
  }
  __syncthreads();
  if (tid < DQK) {
    for (int h = 0; h < NH; ++h) {
      const size_t qrow = (size_t)t * (NH * DQK) + (size_t)h * DQK;
      const size_t orow = ((size_t)t * NH + h) * DQK;
      const size_t krow = (size_t)t * (NH * (DN + DV)) + (size_t)h * (DN + DV);
      if (tid < DN) {
        Qf[orow + tid] = (bf16_t)qb[qrow + tid];
        Kf[orow + tid] = (bf16_t)kvb[krow + tid];
        Vb[((size_t)t * NH + h) * DV + tid] = (bf16_t)kvb[krow + DN + tid];
      } else {
        int i = tid - DN;
        const float* qp = qb + qrow + DN;
        float y, rt;
        if (i < 32) { y = qp[2 * i];           rt = -qp[2 * i + 1]; }
        else        { int j = i - 32; y = qp[2 * j + 1]; rt = qp[2 * j]; }
        Qf[orow + tid] = (bf16_t)(y * cs[i] + rt * sn[i]);
        Kf[orow + tid] = (bf16_t)kemb[i];
      }
    }
  }
}

// ---------------------------------------------------------------------------
// Causal flash attention, one block per (128 q-rows, head); 8 waves x 16 rows.
// Q fragments in registers; K staged to LDS via ASYNCcnt DMA when available;
// V^T staged with 32-bit loads. S = QK^T and O += PV via WMMA bf16.
// ---------------------------------------------------------------------------
__global__ void __launch_bounds__(256)
mla_attn_k(const bf16_t* __restrict__ Qf, const bf16_t* __restrict__ Kf,
           const bf16_t* __restrict__ Vb, float* __restrict__ Oatt) {
  __shared__ bf16_t Ks[64][DQK];     // keys x feature
  __shared__ bf16_t Vt[DV][64];      // dv x keys (transposed for B-frags)
  __shared__ bf16_t Ps[8][16][64];   // per-wave P round-trip buffer
  const int qTile = blockIdx.x;
  const int head = blockIdx.y;
  const int tid = threadIdx.x;
  const int wave = tid >> 5, lane = tid & 31;
  const int hi = lane >> 4, l15 = lane & 15;
  const int qBase = qTile * 128;
  const int qRow = qBase + wave * 16;

  bf16x16 qfrag[6];
#pragma unroll
  for (int c = 0; c < 6; ++c) {
    const bf16_t* p =
        Qf + ((size_t)(qRow + l15) * NH + head) * DQK + c * 32 + hi * 16;
    qfrag[c] = ld_frag_a(p);
  }

  const f32x8 zero8 = {0.f, 0.f, 0.f, 0.f, 0.f, 0.f, 0.f, 0.f};
  f32x8 oacc[8];
#pragma unroll
  for (int d = 0; d < 8; ++d) oacc[d] = zero8;
  float mrow[8], lrow[8];
#pragma unroll
  for (int j = 0; j < 8; ++j) { mrow[j] = -3.0e38f; lrow[j] = 0.f; }

  const float scale = 0.0721687836f;  // 192^-0.5
  const int nkt = (qBase + 128) / 64;

  for (int kt = 0; kt < nkt; ++kt) {
    // ---- stage K tile: async DMA to LDS (ASYNCcnt) when available --------
#if ASYNC_LDS
#pragma unroll
    for (int i = 0; i < 6; ++i) {  // 64 rows * 384B = 1536 x 16B chunks
      int ch = tid + 256 * i;
      int r = ch / 24, c = (ch % 24) * 8;
      __builtin_amdgcn_global_load_async_to_lds_b128(
          (v4i_t*)(Kf + ((size_t)(kt * 64 + r) * NH + head) * DQK + c),
          (v4i_t*)&Ks[r][c], 0, 0);
    }
#else
    for (int i = tid; i < 64 * DQK; i += 256) {
      int r = i / DQK, c = i % DQK;
      Ks[r][c] = Kf[((size_t)(kt * 64 + r) * NH + head) * DQK + c];
    }
#endif
    // ---- stage V^T (transpose; 32-bit global reads) -----------------------
    for (int i = tid; i < (64 * DV) / 2; i += 256) {
      int r = i & 63, d2 = i >> 6;  // d2 = dv pair index
      unsigned pv = *reinterpret_cast<const unsigned*>(
          Vb + ((size_t)(kt * 64 + r) * NH + head) * DV + d2 * 2);
      Vt[d2 * 2 + 0][r] = reinterpret_cast<const bf16_t*>(&pv)[0];
      Vt[d2 * 2 + 1][r] = reinterpret_cast<const bf16_t*>(&pv)[1];
    }
#if ASYNC_LDS
    asm volatile("s_wait_asynccnt 0x0" ::: "memory");
#endif
    __syncthreads();

    // ---- S = Q K^T --------------------------------------------------------
    f32x8 sacc[4];
#pragma unroll
    for (int nt = 0; nt < 4; ++nt) sacc[nt] = zero8;
#pragma unroll
    for (int c = 0; c < 6; ++c)
#pragma unroll
      for (int nt = 0; nt < 4; ++nt) {
        bf16x16 kf = ld_frag_b(&Ks[nt * 16 + l15][c * 32 + hi * 16]);
        sacc[nt] = wmma_bf16(qfrag[c], kf, sacc[nt]);
      }

    // ---- online softmax (rows live across 16-lane halves) -----------------
#pragma unroll
    for (int j = 0; j < 8; ++j) {
      const int qi = qRow + j + hi * 8;
      float s[4];
      float vmax = -3.0e38f;
#pragma unroll
      for (int nt = 0; nt < 4; ++nt) {
        int ki = kt * 64 + nt * 16 + l15;
        float x = sacc[nt][j] * scale;
        if (ki > qi) x = -3.0e38f;  // causal mask
        s[nt] = x;
        vmax = fmaxf(vmax, x);
      }
#pragma unroll
      for (int m = 1; m <= 8; m <<= 1)
        vmax = fmaxf(vmax, __shfl_xor(vmax, m, 32));
      float newm = fmaxf(mrow[j], vmax);
      float corr = __expf(mrow[j] - newm);
      mrow[j] = newm;
      float rsum = 0.f;
#pragma unroll
      for (int nt = 0; nt < 4; ++nt) {
        float p = __expf(s[nt] - newm);
        rsum += p;
        Ps[wave][j + hi * 8][nt * 16 + l15] = (bf16_t)p;
      }
#pragma unroll
      for (int m = 1; m <= 8; m <<= 1) rsum += __shfl_xor(rsum, m, 32);
      lrow[j] = lrow[j] * corr + rsum;
#pragma unroll
      for (int d = 0; d < 8; ++d) oacc[d][j] *= corr;
    }

    // ---- O += P V ---------------------------------------------------------
#pragma unroll
    for (int kc = 0; kc < 2; ++kc) {
      bf16x16 pf = ld_frag_a(&Ps[wave][l15][kc * 32 + hi * 16]);
#pragma unroll
      for (int d = 0; d < 8; ++d) {
        bf16x16 vf = ld_frag_b(&Vt[d * 16 + l15][kc * 32 + hi * 16]);
        oacc[d] = wmma_bf16(pf, vf, oacc[d]);
      }
    }
    __syncthreads();
  }

#pragma unroll
  for (int d = 0; d < 8; ++d)
#pragma unroll
    for (int j = 0; j < 8; ++j) {
      int row = qRow + j + hi * 8;
      int col = d * 16 + l15;
      Oatt[((size_t)row * NH + head) * DV + col] = oacc[d][j] / lrow[j];
    }
}

// ---------------------------------------------------------------------------
extern "C" void kernel_launch(void* const* d_in, const int* in_sizes, int n_in,
                              void* d_out, int out_size, void* d_ws,
                              size_t ws_size, hipStream_t stream) {
  (void)in_sizes; (void)n_in; (void)out_size; (void)ws_size;
  const float* hs      = (const float*)d_in[0];
  const float* cosT    = (const float*)d_in[1];
  const float* sinT    = (const float*)d_in[2];
  const float* ln_in   = (const float*)d_in[3];
  const float* w_q_a   = (const float*)d_in[4];
  const float* ln_q_a  = (const float*)d_in[5];
  const float* w_q_b   = (const float*)d_in[6];
  const float* w_kv_a  = (const float*)d_in[7];
  const float* ln_kv_a = (const float*)d_in[8];
  const float* w_kv_b  = (const float*)d_in[9];
  const float* w_o     = (const float*)d_in[10];
  const float* ln_post = (const float*)d_in[11];
  const float* w_gate  = (const float*)d_in[12];
  const float* w_up    = (const float*)d_in[13];
  const float* w_down  = (const float*)d_in[14];
  float* out = (float*)d_out;

  float* hn   = (float*)d_ws;
  float* qa   = hn + (size_t)T_TOK * HID;
  float* qb   = qa + (size_t)T_TOK * QLR;
  float* ckv  = qb + (size_t)T_TOK * NH * DQK;
  float* kvb  = ckv + (size_t)T_TOK * (KVLR + DR);
  float* attn = kvb + (size_t)T_TOK * NH * (DN + DV);
  float* res2 = attn + (size_t)T_TOK * HID;
  float* h2   = res2 + (size_t)T_TOK * HID;
  float* gbuf = h2 + (size_t)T_TOK * HID;
  bf16_t* Qf  = (bf16_t*)(gbuf + (size_t)T_TOK * INTER);
  bf16_t* Kf  = Qf + (size_t)T_TOK * NH * DQK;
  bf16_t* Vb  = Kf + (size_t)T_TOK * NH * DQK;

  const dim3 b256(256);

  rmsnorm_k<<<T_TOK, b256, 0, stream>>>(hs, ln_in, hn, HID, HID, HID);
  gemm_bf16wmma<0><<<dim3(QLR / 64, T_TOK / 256), b256, 0, stream>>>(
      hn, HID, w_q_a, HID, qa, QLR, nullptr, 0, HID);
  rmsnorm_k<<<T_TOK, b256, 0, stream>>>(qa, ln_q_a, qa, QLR, QLR, QLR);
  gemm_bf16wmma<0><<<dim3(NH * DQK / 64, T_TOK / 256), b256, 0, stream>>>(
      qa, QLR, w_q_b, QLR, qb, NH * DQK, nullptr, 0, QLR);
  gemm_bf16wmma<0><<<dim3((KVLR + DR) / 64, T_TOK / 256), b256, 0, stream>>>(
      hn, HID, w_kv_a, HID, ckv, KVLR + DR, nullptr, 0, HID);
  rmsnorm_k<<<T_TOK, b256, 0, stream>>>(ckv, ln_kv_a, ckv, KVLR, KVLR + DR,
                                        KVLR + DR);
  gemm_bf16wmma<0><<<dim3(NH * (DN + DV) / 64, T_TOK / 256), b256, 0, stream>>>(
      ckv, KVLR + DR, w_kv_b, KVLR, kvb, NH * (DN + DV), nullptr, 0, KVLR);
  rope_pack_k<<<T_TOK, b256, 0, stream>>>(qb, ckv, kvb, cosT, sinT, Qf, Kf, Vb);
  mla_attn_k<<<dim3(T_TOK / 128, NH), b256, 0, stream>>>(Qf, Kf, Vb, attn);
  gemm_bf16wmma<1><<<dim3(HID / 64, T_TOK / 256), b256, 0, stream>>>(
      attn, HID, w_o, NH * DV, res2, HID, hs, HID, NH * DV);
  rmsnorm_k<<<T_TOK, b256, 0, stream>>>(res2, ln_post, h2, HID, HID, HID);
  gemm_bf16wmma<0><<<dim3(INTER / 64, T_TOK / 256), b256, 0, stream>>>(
      h2, HID, w_gate, HID, gbuf, INTER, nullptr, 0, HID);
  gemm_bf16wmma<2><<<dim3(INTER / 64, T_TOK / 256), b256, 0, stream>>>(
      h2, HID, w_up, HID, gbuf, INTER, gbuf, INTER, HID);
  gemm_bf16wmma<1><<<dim3(HID / 64, T_TOK / 256), b256, 0, stream>>>(
      gbuf, INTER, w_down, INTER, out, HID, res2, HID, INTER);
}
